// _MPNNNet_63900523430414
// MI455X (gfx1250) — matrix operand order repaired and checked by reading, not compiled, not run on hardware
//
#include <hip/hip_runtime.h>
#include <hip/hip_bf16.h>
#include <math.h>

// Problem constants (from reference)
#define NN 8192
#define EE 16384
#define GG 256
#define HH 64
#define LL 3

typedef __attribute__((ext_vector_type(16))) __bf16 v16bf;
typedef __attribute__((ext_vector_type(8)))  float  v8f;
typedef int v4i_ __attribute__((vector_size(16)));

// ---------------- async copy to LDS (guarded) ----------------
#if defined(__AMDGCN__) && __has_builtin(__builtin_amdgcn_global_load_async_to_lds_b128)
#define ASYNC_CP 1
#endif

__device__ __forceinline__ void cp_b128(const void* g, void* l) {
#ifdef ASYNC_CP
  __builtin_amdgcn_global_load_async_to_lds_b128(
      (__attribute__((address_space(1))) v4i_*)(void*)g,
      (__attribute__((address_space(3))) v4i_*)(void*)l, 0, 0);
#else
  *(uint4*)l = *(const uint4*)g;
#endif
}

template <int Ncnt>
__device__ __forceinline__ void cp_wait() {
#ifdef ASYNC_CP
#if __has_builtin(__builtin_amdgcn_s_wait_asynccnt)
  __builtin_amdgcn_s_wait_asynccnt((short)Ncnt);
#else
  asm volatile("s_wait_asynccnt %0" ::"i"(Ncnt) : "memory");
#endif
#endif
}

__device__ __forceinline__ float gelu_f(float v) {
  return 0.5f * v * (1.0f + erff(v * 0.70710678118654752f));
}

// ---------------- small utility kernels ----------------
__global__ void k_zero(float* __restrict__ p, int n) {
  int i = blockIdx.x * 256 + threadIdx.x;
  if (i < n) p[i] = 0.0f;
}

__global__ void k_deg(const int* __restrict__ ei, float* __restrict__ deg, int E_) {
  int e = blockIdx.x * 256 + threadIdx.x;
  if (e < E_) atomicAdd(&deg[ei[E_ + e]], 1.0f);
}

__global__ void k_node_embed(const float* __restrict__ x, const float* __restrict__ w,
                             const float* __restrict__ b, float* __restrict__ h, int N_) {
  int idx = blockIdx.x * 256 + threadIdx.x;
  if (idx >= N_ * HH) return;
  int i = idx >> 6, o = idx & 63;
  float a = b[o];
#pragma unroll
  for (int f = 0; f < 9; ++f) a += x[i * 9 + f] * w[f * HH + o];
  h[idx] = gelu_f(a);
}

__global__ void k_edge_mlp(const float* __restrict__ ea, const float* __restrict__ w1,
                           const float* __restrict__ b1, float* __restrict__ ebuf,
                           int E_, int layer) {
  int idx = blockIdx.x * 256 + threadIdx.x;
  if (idx >= E_ * HH) return;
  int e = idx >> 6, o = idx & 63;
  const float* w = w1 + layer * 4 * HH;
  float a = b1[layer * HH + o];
#pragma unroll
  for (int f = 0; f < 4; ++f) a += ea[e * 4 + f] * w[f * HH + o];
  ebuf[idx] = gelu_f(a);
}

// ---------------- pre-swizzle edge_w2 / edge_b2 into WMMA-B bf16 layout -------
// Layout: flat = (((l*130 + c)*4 + t)*32 + L)*16 + p
//   chunk c<128:  k0=c>>1, i=(c&1)*32 + jj ; c in {128,129}: bias, i=(c-128)*32+jj
//   jj = p<8 ? (L/16)*8 + p : 16 + (L/16)*8 + (p-8)  (B operand per-lane K index), n = t*16 + (L%16)
__global__ void k_prep_w2(const float* __restrict__ w2, const float* __restrict__ b2,
                          __bf16* __restrict__ Ball) {
  int idx = blockIdx.x * 256 + threadIdx.x;
  if (idx >= LL * 130 * 2048) return;
  int p = idx & 15;
  int Ln = (idx >> 4) & 31;
  int t  = (idx >> 9) & 3;
  int c  = (idx >> 11) % 130;
  int l  = idx / (130 * 2048);
  int o  = t * 16 + (Ln & 15);
  int jj = (p < 8) ? (Ln >> 4) * 8 + p : 16 + (Ln >> 4) * 8 + (p - 8);
  float val;
  if (c < 128) {
    int k0 = c >> 1;
    int i  = (c & 1) * 32 + jj;
    val = w2[(l * 64 + k0) * 4096 + i * 64 + o];
  } else {
    int i = (c - 128) * 32 + jj;
    val = b2[l * 4096 + i * 64 + o];
  }
  Ball[idx] = (__bf16)val;
}

// ---------------- the heavy fused per-edge GEMM (WMMA) ----------------
// msg[16 edges, 64] = P @ W2' + h_src @ B2, atomically scattered into agg[dst].
#define GWAVES 4
__global__ __launch_bounds__(GWAVES * 32) void k_edge_gemm(
    const float* __restrict__ h, const float* __restrict__ e_buf,
    const __bf16* __restrict__ Ball, const int* __restrict__ edge_index,
    float* __restrict__ agg, int E_, int layer) {
  __shared__ __bf16 sB[2][4096];          // double-buffered B chunk-pair (8KB each)
  __shared__ float  sE[GWAVES][16 * 64];  // per-wave e tile

  const int tid  = threadIdx.x;
  const int lane = tid & 31;
  const int wave = tid >> 5;
  const int ebase = (blockIdx.x * GWAVES + wave) * 16;
  const int mlane = lane & 15;
  const int half  = lane >> 4;          // 0 or 1

  // stage this wave's e tile (16x64 f32) into LDS (wave-private, no barrier needed)
  {
    const float4* es = (const float4*)(e_buf + (size_t)ebase * 64);
    float4* sd = (float4*)sE[wave];
#pragma unroll
    for (int r = 0; r < 8; ++r) sd[r * 32 + lane] = es[r * 32 + lane];
  }

  // per-lane h_src slice in the exact WMMA-A register layout
  const int sidx = edge_index[ebase + mlane];
  const float4* hrow = (const float4*)(h + (size_t)sidx * 64);
  const int hb4 = half * 2;  // (half*8)/4
  float hfA[16], hfB[16];
  {
    float4 a0 = hrow[hb4 + 0],  a1 = hrow[hb4 + 1];
    float4 a2 = hrow[hb4 + 4],  a3 = hrow[hb4 + 5];
    float4 b0 = hrow[hb4 + 8],  b1 = hrow[hb4 + 9];
    float4 b2v = hrow[hb4 + 12], b3 = hrow[hb4 + 13];
    hfA[0]=a0.x; hfA[1]=a0.y; hfA[2]=a0.z; hfA[3]=a0.w;
    hfA[4]=a1.x; hfA[5]=a1.y; hfA[6]=a1.z; hfA[7]=a1.w;
    hfA[8]=a2.x; hfA[9]=a2.y; hfA[10]=a2.z; hfA[11]=a2.w;
    hfA[12]=a3.x; hfA[13]=a3.y; hfA[14]=a3.z; hfA[15]=a3.w;
    hfB[0]=b0.x; hfB[1]=b0.y; hfB[2]=b0.z; hfB[3]=b0.w;
    hfB[4]=b1.x; hfB[5]=b1.y; hfB[6]=b1.z; hfB[7]=b1.w;
    hfB[8]=b2v.x; hfB[9]=b2v.y; hfB[10]=b2v.z; hfB[11]=b2v.w;
    hfB[12]=b3.x; hfB[13]=b3.y; hfB[14]=b3.z; hfB[15]=b3.w;
  }

  // destination node of each accumulator row this lane holds
  int d8[8];
#pragma unroll
  for (int r = 0; r < 8; ++r) d8[r] = edge_index[E_ + ebase + half * 8 + r];

  v8f acc[4];
#pragma unroll
  for (int t = 0; t < 4; ++t) acc[t] = (v8f)0.0f;

  const __bf16* Bl = Ball + (size_t)layer * 130 * 2048;
  // prefetch chunk-pair 0
#pragma unroll
  for (int j = 0; j < 4; ++j)
    cp_b128(Bl + j * 1024 + tid * 8, &sB[0][j * 1024 + tid * 8]);

  const float* sEw = sE[wave];
#pragma unroll 1
  for (int kk = 0; kk < 65; ++kk) {
    if (kk < 64) {
      const __bf16* src = Bl + (size_t)(kk + 1) * 4096;
      __bf16* dst = sB[(kk + 1) & 1];
#pragma unroll
      for (int j = 0; j < 4; ++j)
        cp_b128(src + j * 1024 + tid * 8, dst + j * 1024 + tid * 8);
      cp_wait<4>();
    } else {
      cp_wait<0>();
    }
    __syncthreads();

    const float s = (kk < 64) ? sEw[mlane * 64 + kk] : 1.0f;
    v16bf aE, aO;
#pragma unroll
    for (int p = 0; p < 16; ++p) {
      aE[p] = (__bf16)(s * hfA[p]);
      aO[p] = (__bf16)(s * hfB[p]);
    }
    const __bf16* bp = sB[kk & 1];
#pragma unroll
    for (int t = 0; t < 4; ++t) {
      v16bf bE = *(const v16bf*)(bp + t * 512 + lane * 16);
      v16bf bO = *(const v16bf*)(bp + 2048 + t * 512 + lane * 16);
      acc[t] = __builtin_amdgcn_wmma_f32_16x16x32_bf16(false, aE, false, bE,
                                                       (short)0, acc[t], false, false);
      acc[t] = __builtin_amdgcn_wmma_f32_16x16x32_bf16(false, aO, false, bO,
                                                       (short)0, acc[t], false, false);
    }
    __syncthreads();
  }

  // scatter-add msg rows into agg[dst]
#pragma unroll
  for (int t = 0; t < 4; ++t) {
    const int o = t * 16 + mlane;
#pragma unroll
    for (int r = 0; r < 8; ++r)
      atomicAdd(&agg[(size_t)d8[r] * 64 + o], acc[t][r]);
  }
}

// ---------------- per-node post: scatter-mean + root matmul + gelu + residual + LN ---
__global__ __launch_bounds__(64) void k_layer_post(
    const float* __restrict__ h, const float* __restrict__ agg,
    const float* __restrict__ deg, const float* __restrict__ root_w,
    const float* __restrict__ conv_b, const float* __restrict__ ln_g,
    const float* __restrict__ ln_b, float* __restrict__ hout, int layer) {
  const int i = blockIdx.x;
  const int o = threadIdx.x;
  __shared__ float sh[64];
  __shared__ float red[4];
  sh[o] = h[(size_t)i * 64 + o];
  __syncthreads();
  const float* rw = root_w + layer * 4096;
  float a = conv_b[layer * 64 + o];
#pragma unroll
  for (int j = 0; j < 64; ++j) a += sh[j] * rw[j * 64 + o];
  const float degi = fmaxf(deg[i], 1.0f);
  a += agg[(size_t)i * 64 + o] / degi;
  const float z = sh[o] + gelu_f(a);
  float s1 = z, s2 = z * z;
#pragma unroll
  for (int off = 16; off > 0; off >>= 1) {
    s1 += __shfl_down(s1, off);
    s2 += __shfl_down(s2, off);
  }
  if ((threadIdx.x & 31) == 0) {
    red[(threadIdx.x >> 5) * 2 + 0] = s1;
    red[(threadIdx.x >> 5) * 2 + 1] = s2;
  }
  __syncthreads();
  const float S1 = red[0] + red[2], S2 = red[1] + red[3];
  const float mu = S1 * (1.0f / 64.0f);
  const float var = S2 * (1.0f / 64.0f) - mu * mu;
  hout[(size_t)i * 64 + o] =
      (z - mu) * rsqrtf(var + 1e-5f) * ln_g[layer * 64 + o] + ln_b[layer * 64 + o];
}

// ---------------- pooling + head ----------------
__global__ void k_pool(const float* __restrict__ h, const int* __restrict__ batch,
                       float* __restrict__ mol, float* __restrict__ gcnt, int N_) {
  int idx = blockIdx.x * 256 + threadIdx.x;
  if (idx >= N_ * 64) return;
  int i = idx >> 6, o = idx & 63;
  int b = batch[i];
  atomicAdd(&mol[b * 64 + o], h[idx]);
  if (o == 0) atomicAdd(&gcnt[b], 1.0f);
}

__global__ __launch_bounds__(32) void k_head(
    const float* __restrict__ mol, const float* __restrict__ gcnt,
    const float* __restrict__ w1, const float* __restrict__ b1,
    const float* __restrict__ w2, const float* __restrict__ b2,
    float* __restrict__ out) {
  const int g = blockIdx.x;
  const int j = threadIdx.x;  // 32 hidden units
  const float cnt = fmaxf(gcnt[g], 1.0f);
  float a = b1[j];
#pragma unroll
  for (int q = 0; q < 64; ++q) a += (mol[g * 64 + q] / cnt) * w1[q * 32 + j];
  float v = gelu_f(a) * w2[j];
#pragma unroll
  for (int off = 16; off > 0; off >>= 1) v += __shfl_down(v, off);
  if (j == 0) out[g] = v + b2[0];
}

// ---------------- host orchestration ----------------
extern "C" void kernel_launch(void* const* d_in, const int* in_sizes, int n_in,
                              void* d_out, int out_size, void* d_ws, size_t ws_size,
                              hipStream_t stream) {
  (void)in_sizes; (void)n_in; (void)out_size; (void)ws_size;
  const float* x      = (const float*)d_in[0];
  const int*   ei     = (const int*)d_in[1];
  const float* ea     = (const float*)d_in[2];
  const int*   batch  = (const int*)d_in[3];
  const float* atom_w = (const float*)d_in[4];
  const float* atom_b = (const float*)d_in[5];
  const float* ew1    = (const float*)d_in[6];
  const float* eb1    = (const float*)d_in[7];
  const float* ew2    = (const float*)d_in[8];
  const float* eb2    = (const float*)d_in[9];
  const float* root_w = (const float*)d_in[10];
  const float* conv_b = (const float*)d_in[11];
  const float* ln_g   = (const float*)d_in[12];
  const float* ln_b   = (const float*)d_in[13];
  const float* hw1    = (const float*)d_in[14];
  const float* hb1    = (const float*)d_in[15];
  const float* hw2    = (const float*)d_in[16];
  const float* hb2    = (const float*)d_in[17];
  float* out = (float*)d_out;

  float* ws   = (float*)d_ws;
  float* hA   = ws;                  // N*64
  float* hB   = hA + NN * 64;        // N*64
  float* agg  = hB + NN * 64;        // N*64
  float* deg  = agg + NN * 64;       // N
  float* ebuf = deg + NN;            // E*64
  float* mol  = ebuf + EE * 64;      // G*64
  float* gcnt = mol + GG * 64;       // G
  __bf16* Ball = (__bf16*)(gcnt + GG);  // 3*130*2048 bf16

  k_prep_w2<<<(LL * 130 * 2048 + 255) / 256, 256, 0, stream>>>(ew2, eb2, Ball);
  k_zero<<<(NN + 255) / 256, 256, 0, stream>>>(deg, NN);
  k_deg<<<(EE + 255) / 256, 256, 0, stream>>>(ei, deg, EE);
  k_node_embed<<<(NN * 64 + 255) / 256, 256, 0, stream>>>(x, atom_w, atom_b, hA, NN);

  for (int l = 0; l < LL; ++l) {
    k_edge_mlp<<<(EE * 64 + 255) / 256, 256, 0, stream>>>(ea, ew1, eb1, ebuf, EE, l);
    k_zero<<<(NN * 64 + 255) / 256, 256, 0, stream>>>(agg, NN * 64);
    k_edge_gemm<<<EE / (GWAVES * 16), GWAVES * 32, 0, stream>>>(hA, ebuf, Ball, ei, agg, EE, l);
    k_layer_post<<<NN, 64, 0, stream>>>(hA, agg, deg, root_w, conv_b, ln_g, ln_b, hB, l);
    float* t = hA; hA = hB; hB = t;
  }

  k_zero<<<(GG * 64 + GG + 255) / 256, 256, 0, stream>>>(mol, GG * 64 + GG);
  k_pool<<<(NN * 64 + 255) / 256, 256, 0, stream>>>(hA, batch, mol, gcnt, NN);
  k_head<<<GG, 32, 0, stream>>>(mol, gcnt, hw1, hb1, hw2, hb2, out);
}
